// Attention_2465311228566
// MI455X (gfx1250) — matrix operand order repaired and checked by reading, not compiled, not run on hardware
//
#include <hip/hip_runtime.h>
#include <hip/hip_bf16.h>
#include <math.h>

// ---------------------------------------------------------------------------
// Dims (fixed by the reference)
// ---------------------------------------------------------------------------
#define B_   512
#define S_   64
#define H_   1024
#define E_   256
#define V_   128
#define L_   2

typedef __attribute__((ext_vector_type(16))) _Float16 v16h;
typedef __attribute__((ext_vector_type(8)))  _Float16 v8h;
typedef __attribute__((ext_vector_type(2)))  _Float16 v2h;
typedef __attribute__((ext_vector_type(8)))  float    v8f;

union V16 { v16h v; v8h h[2]; };

enum { EPI_STORE = 0, EPI_ACCUM = 1, EPI_SCORE = 2 };

// ---------------------------------------------------------------------------
// Tiled WMMA GEMM:  C[M,N] (+)= A[M,K] * B   (fp32 in, f16 WMMA, f32 accum)
//   TRANSB=false : B is K x N row-major (ldb = N)
//   TRANSB=true  : B is N x K row-major (ldb = K)  (torch x @ W^T)
//   BIAS         : number of bias vectors added in EPI_STORE (0/1/2)
// Block: 256 threads (8 waves). Tile BM=128 x BN=128, BK=32 (one WMMA K-step).
// Waves 4(M) x 2(N); wave tile 32x64 = 2x4 WMMA 16x16 tiles (8 wmma / K-step).
// Software-pipelined: double-buffered LDS, global prefetch overlapped with
// WMMA, one barrier per K-step.
// EPI_SCORE: e[row] += sum_col Vw[col]*tanh(acc + t2[b,col]) via shfl reduce.
// ---------------------------------------------------------------------------
template <int EPI, bool TRANSB, int BIAS>
__global__ __launch_bounds__(256)
void gemm_wmma(const float* __restrict__ A, int lda,
               const float* __restrict__ Bm, int ldb,
               float* __restrict__ C, int ldc, int K,
               const float* __restrict__ bias1, const float* __restrict__ bias2,
               const float* __restrict__ t2, const float* __restrict__ Vw,
               float* __restrict__ e_out)
{
    constexpr int BM = 128, BN = 128, BK = 32;

    __shared__ __align__(16) _Float16 A_s[2][BM * BK];  // [m][k], k-stride 32
    __shared__ __align__(16) _Float16 B_s[2][BN * BK];  // [n][k]

    const int n0 = blockIdx.x * BN;
    const int m0 = blockIdx.y * BM;
    const int t  = threadIdx.x;
    const int lane  = t & 31;
    const int wave  = t >> 5;
    const int waveM = wave >> 1;          // 0..3  (row offset waveM*32)
    const int waveN = wave & 1;           // 0..1  (col offset waveN*64)
    const int hi    = lane >> 4;          // lane half (ISA fragment layout)
    const int l16   = lane & 15;

    v8f acc[2][4];
    #pragma unroll
    for (int a = 0; a < 2; ++a)
        #pragma unroll
        for (int b = 0; b < 4; ++b)
            #pragma unroll
            for (int v = 0; v < 8; ++v) acc[a][b][v] = 0.0f;

    // per-thread staging coordinates (8 packed dwords for A tile, 8 for B tile)
    float2 ra[8], rb[8];

    auto load_global = [&](int k0) {
        #pragma unroll
        for (int i = 0; i < 8; ++i) {
            int e2 = t + i * 256;
            int m = e2 >> 4, kp = e2 & 15;
            ra[i] = *(const float2*)&A[(size_t)(m0 + m) * lda + k0 + 2 * kp];
        }
        if constexpr (TRANSB) {
            #pragma unroll
            for (int i = 0; i < 8; ++i) {
                int e2 = t + i * 256;
                int n = e2 >> 4, kp = e2 & 15;          // contiguous k in global
                rb[i] = *(const float2*)&Bm[(size_t)(n0 + n) * ldb + k0 + 2 * kp];
            }
        } else {
            #pragma unroll
            for (int i = 0; i < 8; ++i) {
                int e2 = t + i * 256;
                int kp = e2 >> 7, n = e2 & 127;         // contiguous n in global
                rb[i].x = Bm[(size_t)(k0 + 2 * kp) * ldb + n0 + n];
                rb[i].y = Bm[(size_t)(k0 + 2 * kp + 1) * ldb + n0 + n];
            }
        }
    };

    auto store_lds = [&](int buf) {
        #pragma unroll
        for (int i = 0; i < 8; ++i) {
            int e2 = t + i * 256;
            int m = e2 >> 4, kp = e2 & 15;
            v2h h; h.x = (_Float16)ra[i].x; h.y = (_Float16)ra[i].y;
            *(v2h*)&A_s[buf][m * 32 + 2 * kp] = h;
        }
        #pragma unroll
        for (int i = 0; i < 8; ++i) {
            int e2 = t + i * 256;
            int n, kp;
            if constexpr (TRANSB) { n = e2 >> 4; kp = e2 & 15; }
            else                  { kp = e2 >> 7; n = e2 & 127; }
            v2h h; h.x = (_Float16)rb[i].x; h.y = (_Float16)rb[i].y;
            *(v2h*)&B_s[buf][n * 32 + 2 * kp] = h;
        }
    };

    // ---- pipeline prologue ----
    load_global(0);
    store_lds(0);
    __syncthreads();

    const int nIter = K / BK;
    for (int it = 0; it < nIter; ++it) {
        const int cur = it & 1;

        // prefetch next K-step from global while this step computes
        if (it + 1 < nIter) load_global((it + 1) * BK);

        // ---- fragments (ISA 7.12.2 layouts) ----
        V16 afrag[2], bfrag[4];
        #pragma unroll
        for (int tm = 0; tm < 2; ++tm) {
            int ml = waveM * 32 + tm * 16 + l16;
            afrag[tm].h[0] = *(const v8h*)&A_s[cur][ml * 32 + 8 * hi];
            afrag[tm].h[1] = *(const v8h*)&A_s[cur][ml * 32 + 16 + 8 * hi];
        }
        #pragma unroll
        for (int tn = 0; tn < 4; ++tn) {
            int nl = waveN * 64 + tn * 16 + l16;
            bfrag[tn].h[0] = *(const v8h*)&B_s[cur][nl * 32 + 8 * hi];
            bfrag[tn].h[1] = *(const v8h*)&B_s[cur][nl * 32 + 16 + 8 * hi];
        }
        #pragma unroll
        for (int tm = 0; tm < 2; ++tm)
            #pragma unroll
            for (int tn = 0; tn < 4; ++tn)
                acc[tm][tn] = __builtin_amdgcn_wmma_f32_16x16x32_f16(
                    false, afrag[tm].v, false, bfrag[tn].v,
                    (short)0, acc[tm][tn], false, false);

        // stage next step into the other LDS buffer (written last read at it-1)
        if (it + 1 < nIter) store_lds(cur ^ 1);
        __syncthreads();
    }

    // ------------------------------- epilogues ------------------------------
    if constexpr (EPI == EPI_SCORE) {
        // wave's 32 rows lie inside one batch element (m0 % 128 == 0, S == 64)
        const int b = (m0 + waveM * 32) / S_;
        float vw4[4], t24[4];
        #pragma unroll
        for (int tn = 0; tn < 4; ++tn) {
            int col = n0 + waveN * 64 + tn * 16 + l16;
            vw4[tn] = Vw[col];
            t24[tn] = t2[(size_t)b * H_ + col];
        }
        #pragma unroll
        for (int tm = 0; tm < 2; ++tm)
            #pragma unroll
            for (int v = 0; v < 8; ++v) {
                float p = 0.0f;
                #pragma unroll
                for (int tn = 0; tn < 4; ++tn)
                    p += vw4[tn] * tanhf(acc[tm][tn][v] + t24[tn]);
                // reduce over the 16 columns held by lanes l16=0..15 (wave32)
                p += __shfl_xor(p, 1);
                p += __shfl_xor(p, 2);
                p += __shfl_xor(p, 4);
                p += __shfl_xor(p, 8);
                if (l16 == 0)
                    atomicAdd(&e_out[m0 + waveM * 32 + tm * 16 + v + 8 * hi], p);
            }
    } else {
        float bv[4];
        #pragma unroll
        for (int tn = 0; tn < 4; ++tn) {
            int col = n0 + waveN * 64 + tn * 16 + l16;
            float b = 0.0f;
            if constexpr (BIAS >= 1) b += bias1[col];
            if constexpr (BIAS >= 2) b += bias2[col];
            bv[tn] = b;
        }
        #pragma unroll
        for (int tm = 0; tm < 2; ++tm)
            #pragma unroll
            for (int tn = 0; tn < 4; ++tn)
                #pragma unroll
                for (int v = 0; v < 8; ++v) {
                    size_t row = m0 + waveM * 32 + tm * 16 + v + 8 * hi;
                    int    col = n0 + waveN * 64 + tn * 16 + l16;
                    if constexpr (EPI == EPI_ACCUM)
                        C[row * ldc + col] += acc[tm][tn][v];
                    else
                        C[row * ldc + col] = acc[tm][tn][v] + bv[tn];
                }
    }
}

// ---------------------------------------------------------------------------
// Small helper kernels
// ---------------------------------------------------------------------------
__global__ void k_zero(float* p, int n) {
    int i = blockIdx.x * blockDim.x + threadIdx.x;
    if (i < n) p[i] = 0.0f;
}

// x[b, 0:E] = emb[input_ids[b]] ; one block per b, 256 threads = E
__global__ void k_embed(const int* __restrict__ ids, const float* __restrict__ emb,
                        float* __restrict__ x)
{
    int b = blockIdx.x;
    x[(size_t)b * (E_ + H_) + threadIdx.x] = emb[(size_t)ids[b] * E_ + threadIdx.x];
}

// per-b softmax over S, then context ct = sum_s a[s]*enc[b,s,:] -> x[b, E:E+H]
__global__ __launch_bounds__(256)
void k_softmax_ctx(const float* __restrict__ e, const float* __restrict__ enc,
                   float* __restrict__ x)
{
    __shared__ float a_s[S_];
    int b = blockIdx.x, t = threadIdx.x;
    if (t < S_) a_s[t] = e[(size_t)b * S_ + t];
    __syncthreads();
    if (t == 0) {
        float m = -1e30f;
        for (int s = 0; s < S_; ++s) m = fmaxf(m, a_s[s]);
        float sum = 0.0f;
        for (int s = 0; s < S_; ++s) { a_s[s] = expf(a_s[s] - m); sum += a_s[s]; }
        float inv = 1.0f / sum;
        for (int s = 0; s < S_; ++s) a_s[s] *= inv;
    }
    __syncthreads();
    const float* encb = enc + (size_t)b * S_ * H_;
    for (int h = t; h < H_; h += 256) {
        float acc = 0.0f;
        #pragma unroll 8
        for (int s = 0; s < S_; ++s) acc += a_s[s] * encb[(size_t)s * H_ + h];
        x[(size_t)b * (E_ + H_) + E_ + h] = acc;
    }
}

// LSTM pointwise: gates (B,4H) order i,f,g,o -> h_out, c_out
__global__ void k_lstm_cell(const float* __restrict__ gates, const float* __restrict__ c_in,
                            float* __restrict__ h_out, float* __restrict__ c_out)
{
    int idx = blockIdx.x * blockDim.x + threadIdx.x;   // over B*H
    int b = idx / H_, j = idx % H_;
    size_t base = (size_t)b * 4 * H_;
    float gi = gates[base + j];
    float gf = gates[base + H_ + j];
    float gg = gates[base + 2 * H_ + j];
    float go = gates[base + 3 * H_ + j];
    float si = 1.0f / (1.0f + expf(-gi));
    float sf = 1.0f / (1.0f + expf(-gf));
    float so = 1.0f / (1.0f + expf(-go));
    float c2 = sf * c_in[(size_t)b * H_ + j] + si * tanhf(gg);
    h_out[(size_t)b * H_ + j] = so * tanhf(c2);
    c_out[(size_t)b * H_ + j] = c2;
}

// ---------------------------------------------------------------------------
// Launch
// ---------------------------------------------------------------------------
extern "C" void kernel_launch(void* const* d_in, const int* in_sizes, int n_in,
                              void* d_out, int out_size, void* d_ws, size_t ws_size,
                              hipStream_t stream)
{
    const int*   ids   = (const int*)  d_in[0];
    const float* hid   = (const float*)d_in[1];   // (L,B,H)
    const float* cell  = (const float*)d_in[2];   // (L,B,H)
    const float* enc   = (const float*)d_in[3];   // (B,S,H)
    const float* emb   = (const float*)d_in[4];   // (V,E)
    const float* U     = (const float*)d_in[5];   // (H,H)
    const float* W     = (const float*)d_in[6];   // (H,H)
    const float* Vw    = (const float*)d_in[7];   // (H,1)
    const float* Wih0  = (const float*)d_in[8];   // (4H, E+H)
    const float* Whh0  = (const float*)d_in[9];   // (4H, H)
    const float* bih0  = (const float*)d_in[10];
    const float* bhh0  = (const float*)d_in[11];
    const float* Wih1  = (const float*)d_in[12];  // (4H, H)
    const float* Whh1  = (const float*)d_in[13];
    const float* bih1  = (const float*)d_in[14];
    const float* bhh1  = (const float*)d_in[15];
    const float* out_w = (const float*)d_in[16];  // (V,H)
    const float* out_b = (const float*)d_in[17];

    const float* hid0 = hid;                     // layer 0 h
    const float* hid1 = hid + (size_t)B_ * H_;   // layer 1 (top) h
    const float* cel0 = cell;
    const float* cel1 = cell + (size_t)B_ * H_;

    // workspace layout (floats)
    float* ws    = (float*)d_ws;
    float* t2    = ws;                                   // B*H
    float* e     = t2 + (size_t)B_ * H_;                 // B*S
    float* x     = e + (size_t)B_ * S_;                  // B*(E+H)
    float* gates = x + (size_t)B_ * (E_ + H_);           // B*4H (reused both layers)

    // output layout: out (B*V) | h_new (2*B*H) | c_new (2*B*H)
    float* outp  = (float*)d_out;
    float* h_new = outp + (size_t)B_ * V_;
    float* c_new = h_new + (size_t)2 * B_ * H_;
    float* h0o = h_new;
    float* h1o = h_new + (size_t)B_ * H_;
    float* c0o = c_new;
    float* c1o = c_new + (size_t)B_ * H_;

    dim3 blk(256);

    // 1) embedding gather into x[:, :E]
    k_embed<<<B_, E_, 0, stream>>>(ids, emb, x);

    // 2) t2 = h_top @ W          (512x1024)(1024x1024), B normal K x N
    gemm_wmma<EPI_STORE, false, 0><<<dim3(H_ / 128, B_ / 128), blk, 0, stream>>>(
        hid1, H_, W, H_, t2, H_, H_, nullptr, nullptr, nullptr, nullptr, nullptr);

    // 3) e = 0 ; fused t1-GEMM + tanh + Vw reduction (t1 never materialized)
    k_zero<<<(B_ * S_ + 255) / 256, blk, 0, stream>>>(e, B_ * S_);
    gemm_wmma<EPI_SCORE, false, 0><<<dim3(H_ / 128, (B_ * S_) / 128), blk, 0, stream>>>(
        enc, H_, U, H_, nullptr, 0, H_, nullptr, nullptr, t2, Vw, e);

    // 4) softmax over S + context into x[:, E:]
    k_softmax_ctx<<<B_, blk, 0, stream>>>(e, enc, x);

    // 5) LSTM layer 0: gates = x @ Wih0^T + h0 @ Whh0^T + biases
    gemm_wmma<EPI_STORE, true, 2><<<dim3(4 * H_ / 128, B_ / 128), blk, 0, stream>>>(
        x, E_ + H_, Wih0, E_ + H_, gates, 4 * H_, E_ + H_, bih0, bhh0,
        nullptr, nullptr, nullptr);
    gemm_wmma<EPI_ACCUM, true, 0><<<dim3(4 * H_ / 128, B_ / 128), blk, 0, stream>>>(
        hid0, H_, Whh0, H_, gates, 4 * H_, H_, nullptr, nullptr,
        nullptr, nullptr, nullptr);
    k_lstm_cell<<<(B_ * H_) / 256, blk, 0, stream>>>(gates, cel0, h0o, c0o);

    // 6) LSTM layer 1
    gemm_wmma<EPI_STORE, true, 2><<<dim3(4 * H_ / 128, B_ / 128), blk, 0, stream>>>(
        h0o, H_, Wih1, H_, gates, 4 * H_, H_, bih1, bhh1, nullptr, nullptr, nullptr);
    gemm_wmma<EPI_ACCUM, true, 0><<<dim3(4 * H_ / 128, B_ / 128), blk, 0, stream>>>(
        hid1, H_, Whh1, H_, gates, 4 * H_, H_, nullptr, nullptr,
        nullptr, nullptr, nullptr);
    k_lstm_cell<<<(B_ * H_) / 256, blk, 0, stream>>>(gates, cel1, h1o, c1o);

    // 7) vocab projection: out = h1 @ out_w^T + out_b   (N = V = 128 -> 1 n-tile)
    gemm_wmma<EPI_STORE, true, 1><<<dim3(V_ / 128, B_ / 128), blk, 0, stream>>>(
        h1o, H_, out_w, H_, outp, V_, H_, out_b, nullptr, nullptr, nullptr, nullptr);
}
